// SELFYBlock_65755949301933
// MI455X (gfx1250) — compile-verified, matrix operand order checked by reading, not run on hardware
//
#include <hip/hip_runtime.h>
#include <hip/hip_bf16.h>
#include <math.h>

// ---------------- problem constants ----------------
#define H_    14
#define W_    14
#define LPOS  196          // H*W
#define TT    8            // T_SEG
#define BB    16           // batch b
#define BT    128          // b*T
#define DD    768          // model dim
#define DHC   64           // hidden dim / conv channels
#define LW    5            // temporal window
#define UU    9            // spatial window
#define NCH   81           // U*U correlation channels
#define NCHP  96           // padded to multiple of 32 for WMMA K
#define CC    64
#define NTOK  (LPOS*BT)    // 25088 tokens
#define POSP  208          // 196 padded to 13*16
#define SP    (TT*LPOS)    // 1568 spatial elements per (batch) tensor plane group

// ---------------- vector types ----------------
typedef _Float16     half16 __attribute__((ext_vector_type(16)));
typedef float        float8 __attribute__((ext_vector_type(8)));
typedef unsigned int u32x4  __attribute__((ext_vector_type(4)));
typedef int          i32x4  __attribute__((ext_vector_type(4)));
typedef int          i32x8  __attribute__((ext_vector_type(8)));

// ---------------- WMMA helpers (CDNA5 wave32, 16x16x32 f16 -> f32) ----------------
// A/B fragment layout (ISA 7.12.2, 16-bit 16x32): lane L holds row (L&15);
// halves [0..7] = K[(L>>4)*8 .. +7], halves [8..15] = K[16+(L>>4)*8 .. +7].
__device__ __forceinline__ half16 frag_ld(const _Float16* base, int ldk) {
  const int lane = threadIdx.x & 31;
  const _Float16* r = base + (size_t)(lane & 15) * ldk + ((lane >> 4) << 3);
  union { half16 v; float4 q[2]; } u;
  u.q[0] = *(const float4*)(r);
  u.q[1] = *(const float4*)(r + 16);
  return u.v;
}
// Same but caller supplies per-lane row base pointer (for implicit-GEMM gathers).
__device__ __forceinline__ half16 frag_ld_ptr(const _Float16* rowbase) {
  const int lane = threadIdx.x & 31;
  const _Float16* r = rowbase + ((lane >> 4) << 3);
  union { half16 v; float4 q[2]; } u;
  u.q[0] = *(const float4*)(r);
  u.q[1] = *(const float4*)(r + 16);
  return u.v;
}
__device__ __forceinline__ float8 wmma32(half16 a, half16 b, float8 c) {
  return __builtin_amdgcn_wmma_f32_16x16x32_f16(false, a, false, b, (short)0, c,
                                                false, false);
}
__device__ __forceinline__ float bn_gelu(float v, float sc, float sh) {
  float z = v * sc + sh;
  return 0.5f * z * (1.0f + erff(z * 0.70710678118f));
}

// ---------------- Tensor Data Mover (gfx1250): async 2D fp16 tile -> LDS ----------
#if defined(__HIP_DEVICE_COMPILE__)
__device__ __forceinline__ void tdm_load_2d(unsigned lds_off, const void* gptr,
                                            unsigned tile_d0, unsigned tile_d1,
                                            unsigned tensor_d0,
                                            unsigned long long stride0 /*elems*/) {
  unsigned long long ga = (unsigned long long)gptr;
  // D# group 0: count=1, lds_addr, global_addr[56:0], type=2 ("image")
  u32x4 g0;
  g0[0] = 1u;
  g0[1] = lds_off;
  g0[2] = (unsigned)(ga & 0xFFFFFFFFull);
  g0[3] = (unsigned)((ga >> 32) & 0x01FFFFFFull) | (2u << 30);
  // D# group 1: data_size=1 (2B), tensor_dim0/1, tile_dim0/1, dim0 stride
  i32x8 g1;
  g1[0] = (int)(1u << 16);                                            // data_size=2B
  g1[1] = (int)((tensor_d0 & 0xFFFFu) << 16);                         // tdim0 lo
  g1[2] = (int)(((tensor_d0 >> 16) & 0xFFFFu) | ((tile_d1 & 0xFFFFu) << 16)); // tdim0 hi | tdim1 lo
  g1[3] = (int)(((tile_d1 >> 16) & 0xFFFFu) | ((tile_d0 & 0xFFFFu) << 16));   // tdim1 hi | tile_dim0
  g1[4] = (int)(tile_d1 & 0xFFFFu);                                   // tile_dim1, tile_dim2=0
  g1[5] = (int)(stride0 & 0xFFFFFFFFull);                             // dim0 stride lo
  g1[6] = (int)((stride0 >> 32) & 0xFFFFull);                         // dim0 stride hi
  g1[7] = 0;
  i32x4 z4 = {0, 0, 0, 0};
#if __clang_major__ >= 23
  i32x8 z8 = {0, 0, 0, 0, 0, 0, 0, 0};
  __builtin_amdgcn_tensor_load_to_lds(g0, g1, z4, z4, z8, 0);
#else
  __builtin_amdgcn_tensor_load_to_lds(g0, g1, z4, z4, 0);
#endif
}
#endif

// =====================================================================
// Kernel 1: LayerNorm(D=768) + input projection GEMM (25088x768 @ 768x64)
// Block = 128 thr (4 waves); each block does a 64-token x 64-out tile.
// =====================================================================
__global__ void k_ln_proj(const float* __restrict__ x, const float* __restrict__ gamma,
                          const float* __restrict__ beta, const float* __restrict__ in_w,
                          const float* __restrict__ in_b, float* __restrict__ h) {
  __shared__ __align__(16) _Float16 lA[64 * 32];
  __shared__ __align__(16) _Float16 lBt[64 * 32];
  __shared__ float redS[128];
  __shared__ float muS[64];
  __shared__ float rsS[64];

  const int tid = threadIdx.x;
  const int wave = tid >> 5;
  const int lane = tid & 31;
  const int tok0 = blockIdx.x * 64;

  // ---- LN stats: 2 threads per token ----
  {
    const int tl = tid >> 1;
    const int hf = tid & 1;
    const float* row = x + (size_t)(tok0 + tl) * DD + hf * (DD / 2);
    float s = 0.f, s2 = 0.f;
    for (int i = 0; i < DD / 2; ++i) { float v = row[i]; s += v; s2 += v * v; }
    redS[tid] = s;
    __syncthreads();
    if (hf == 0) muS[tl] = (redS[tid] + redS[tid + 1]) * (1.0f / DD);
    __syncthreads();
    redS[tid] = s2;
    __syncthreads();
    if (hf == 0) {
      float mu = muS[tl];
      float var = (redS[tid] + redS[tid + 1]) * (1.0f / DD) - mu * mu;
      rsS[tl] = rsqrtf(var + 1e-5f);
    }
    __syncthreads();
  }

  float8 acc[4] = {};
  for (int kc = 0; kc < DD / 32; ++kc) {
    __syncthreads();
    for (int i = 0; i < 16; ++i) {
      int idx = tid * 16 + i;
      int rl = idx >> 5, kk = idx & 31;
      int kg = kc * 32 + kk;
      float v = x[(size_t)(tok0 + rl) * DD + kg];
      v = (v - muS[rl]) * rsS[rl] * gamma[kg] + beta[kg];
      lA[rl * 32 + kk] = (_Float16)v;
      lBt[rl * 32 + kk] = (_Float16)in_w[(size_t)rl * DD + kg];
    }
    if (kc + 1 < DD / 32)
      __builtin_prefetch(x + (size_t)(tok0 + (tid >> 1)) * DD + (kc + 1) * 32, 0, 3);
    __syncthreads();
    half16 a = frag_ld(lA + wave * 16 * 32, 32);
    for (int nt = 0; nt < 4; ++nt) {
      half16 b = frag_ld(lBt + nt * 16 * 32, 32);
      acc[nt] = wmma32(a, b, acc[nt]);
    }
  }
  const int hiLane = (lane >> 4) << 3;
  for (int nt = 0; nt < 4; ++nt)
    for (int r = 0; r < 8; ++r) {
      int tok = tok0 + wave * 16 + r + hiLane;
      int n = nt * 16 + (lane & 15);
      h[(size_t)tok * DHC + n] = acc[nt][r] + in_b[n];
    }
}

// =====================================================================
// Kernel 2: per-(bt,pos) channel L2 normalize, f32 h -> f16 fn (pos padded to 208)
// =====================================================================
__global__ void k_normalize(const float* __restrict__ h, _Float16* __restrict__ fn) {
  int r = blockIdx.x * blockDim.x + threadIdx.x;
  if (r >= BT * POSP) return;
  int bt = r / POSP, pos = r % POSP;
  _Float16* dst = fn + ((size_t)bt * POSP + pos) * DHC;
  if (pos >= LPOS) {
    for (int c = 0; c < DHC; ++c) dst[c] = (_Float16)0.f;
    return;
  }
  const float* src = h + ((size_t)pos * BT + bt) * DHC;
  float ss = 0.f;
  for (int c = 0; c < DHC; ++c) { float v = src[c]; ss += v * v; }
  float rn = 1.0f / fmaxf(sqrtf(ss), 1e-7f);
  for (int c = 0; c < DHC; ++c) dst[c] = (_Float16)(src[c] * rn);
}

// =====================================================================
// Kernel 3: windowed correlation. Per batch bi=(bb,t,lw): C = Fn_src · Fn_tgtᵀ
// via WMMA row-panels (16 src pos x 208 tgt pos in LDS), then gather 9x9
// windows into s[80][96][8][196] (channels 81..95 zero padded).
// =====================================================================
__global__ void k_corr(const _Float16* __restrict__ fn, _Float16* __restrict__ s) {
  __shared__ float corrS[4][16][POSP];
  const int tid = threadIdx.x, wave = tid >> 5, lane = tid & 31;
  const int bi = blockIdx.x;                 // ((bb*T + t)*Lw + lw)
  const int bb = bi / (TT * LW);
  const int rem = bi % (TT * LW);
  const int t = rem / LW, lw = rem % LW;
  const int st = blockIdx.y * 4 + wave;      // source row-tile
  const int tt = t + lw - 2;
  const bool tv = (tt >= 0) && (tt < TT);
  const bool active = st < 13;

  const _Float16* src = fn + (size_t)(bb * TT + t) * POSP * DHC;
  const _Float16* tgt = fn + (size_t)(bb * TT + (tv ? tt : 0)) * POSP * DHC;

  if (active) {
    const int hiLane = (lane >> 4) << 3;
    for (int nt = 0; nt < 13; ++nt) {
      float8 acc = {};
      if (tv) {
        for (int kc = 0; kc < 2; ++kc) {
          half16 a = frag_ld(src + (size_t)(st * 16) * DHC + kc * 32, DHC);
          half16 b = frag_ld(tgt + (size_t)(nt * 16) * DHC + kc * 32, DHC);
          acc = wmma32(a, b, acc);
        }
      }
      for (int r = 0; r < 8; ++r)
        corrS[wave][r + hiLane][nt * 16 + (lane & 15)] = acc[r];
    }
  }
  __syncthreads();
  if (active) {
    _Float16* sb = s + (size_t)(bb * LW + lw) * NCHP * SP + (size_t)t * LPOS;
    for (int idx = lane; idx < 16 * NCH; idx += 32) {
      int m = idx / NCH, o = idx % NCH;
      int p = st * 16 + m;
      if (p >= LPOS) continue;
      int hh = p / W_, ww = p % W_;
      int u = o / UU, v = o % UU;
      int th = hh + u - 4, tw = ww + v - 4;
      float val = 0.f;
      if (th >= 0 && th < H_ && tw >= 0 && tw < W_) val = corrS[wave][m][th * W_ + tw];
      sb[(size_t)o * SP + p] = (_Float16)val;
    }
    for (int idx = lane; idx < 16 * (NCHP - NCH); idx += 32) {
      int m = idx / (NCHP - NCH), o = NCH + idx % (NCHP - NCH);
      int p = st * 16 + m;
      if (p < LPOS) sb[(size_t)o * SP + p] = (_Float16)0.f;
    }
  }
}

// =====================================================================
// Kernel 4: 1x1x1 "ext" conv as GEMM (125440 x 96pad @ 96 x 64) + BN + GELU
// =====================================================================
__global__ void k_ext(const _Float16* __restrict__ s, const float* __restrict__ w,
                      const float* __restrict__ scale, const float* __restrict__ shift,
                      _Float16* __restrict__ y) {
  __shared__ __align__(16) _Float16 lA[64 * 32];
  __shared__ __align__(16) _Float16 lBt[64 * 32];
  const int tid = threadIdx.x, wave = tid >> 5, lane = tid & 31;
  const int bl = blockIdx.x;
  const int tp0 = blockIdx.y * 64;
  const _Float16* sb = s + (size_t)bl * NCHP * SP;

  float8 acc[4] = {};
  for (int kc = 0; kc < 3; ++kc) {
    __syncthreads();
    for (int i = 0; i < 16; ++i) {
      int idx = tid * 16 + i;
      int rl = idx >> 5, kk = idx & 31;
      int ch = kc * 32 + kk;
      int tp = tp0 + rl;
      lA[rl * 32 + kk] = (tp < SP) ? sb[(size_t)ch * SP + tp] : (_Float16)0.f;
      lBt[rl * 32 + kk] = (ch < NCH) ? (_Float16)w[rl * NCH + ch] : (_Float16)0.f;
    }
    __syncthreads();
    half16 a = frag_ld(lA + wave * 16 * 32, 32);
    for (int nt = 0; nt < 4; ++nt) {
      half16 b = frag_ld(lBt + nt * 16 * 32, 32);
      acc[nt] = wmma32(a, b, acc[nt]);
    }
  }
  const int hiLane = (lane >> 4) << 3;
  for (int nt = 0; nt < 4; ++nt)
    for (int r = 0; r < 8; ++r) {
      int tp = tp0 + wave * 16 + r + hiLane;
      int n = nt * 16 + (lane & 15);
      if (tp < SP)
        y[(size_t)bl * CC * SP + (size_t)n * SP + tp] =
            (_Float16)bn_gelu(acc[nt][r], scale[n], shift[n]);
    }
}

// =====================================================================
// Kernel 5: 3x3 conv as implicit GEMM. Block = one (batch,t) plane.
// Input plane TDM-loaded raw into LDS, rearranged to position-major with halo;
// each wave owns one 16-wide N-tile, keeps 13 M-tile accumulators, K=groups*576.
// groups=1 -> i0/i1 (Cin=64); groups=5 -> i2 (Cin=320, input batches grouped).
// =====================================================================
__global__ void k_conv(const _Float16* __restrict__ X, const float* __restrict__ w,
                       const float* __restrict__ scale, const float* __restrict__ shift,
                       _Float16* __restrict__ Y, int groups) {
  __shared__ __align__(16) _Float16 lRaw[CC * LPOS];  // raw channel-major t-slice
  __shared__ __align__(16) _Float16 lX[256 * CC];     // 16x16 halo, position-major
  __shared__ __align__(16) _Float16 lBt[64 * 32];
  const int tid = threadIdx.x, wave = tid >> 5, lane = tid & 31;
  const int obl = blockIdx.x;
  const int t = blockIdx.y;
  const int cinTot = groups * CC;
  const int nt = wave;

  float8 acc[13] = {};

  for (int g = 0; g < groups; ++g) {
    const _Float16* plane =
        X + (size_t)(obl * groups + g) * CC * SP + (size_t)t * LPOS;
    __syncthreads();
#if defined(__HIP_DEVICE_COMPILE__)
    // Async DMA: 64 rows x 196 f16, row stride SP elems, into lRaw (TDM path).
    if (wave == 0) {
      tdm_load_2d((unsigned)(size_t)&lRaw[0], plane, LPOS, CC, SP,
                  (unsigned long long)SP);
      __builtin_amdgcn_s_wait_tensorcnt(0);
    }
#endif
    __syncthreads();
    for (int idx = tid; idx < 256 * CC; idx += 128) {
      int pos = idx >> 6, ch = idx & 63;
      int hs = (pos >> 4) - 1, ws2 = (pos & 15) - 1;
      _Float16 v = (_Float16)0.f;
      if (hs >= 0 && hs < H_ && ws2 >= 0 && ws2 < W_) v = lRaw[ch * LPOS + hs * W_ + ws2];
      lX[pos * CC + ch] = v;
    }
    for (int kc = 0; kc < 18; ++kc) {
      const int tap = kc >> 1, chal = (kc & 1) * 32;
      const int dh = tap / 3 - 1, dw = tap % 3 - 1;
      __syncthreads();
      for (int i = 0; i < 16; ++i) {
        int idx = tid * 16 + i;
        int n = idx >> 5, kk = idx & 31;
        lBt[n * 32 + kk] =
            (_Float16)w[(size_t)n * cinTot * 9 + (size_t)(g * CC + chal + kk) * 9 + tap];
      }
      __syncthreads();
      half16 b = frag_ld(lBt + nt * 16 * 32, 32);
      for (int mt = 0; mt < 13; ++mt) {
        int m = lane & 15;
        int p = mt * 16 + m;
        if (p >= LPOS) p = LPOS - 1;  // junk lanes, guarded at store
        int hh = p / W_, ww = p % W_;
        const _Float16* rowbase = lX + ((hh + 1 + dh) * 16 + (ww + 1 + dw)) * CC + chal;
        half16 a = frag_ld_ptr(rowbase);
        acc[mt] = wmma32(a, b, acc[mt]);
      }
    }
  }
  const int hiLane = (lane >> 4) << 3;
  const int n = nt * 16 + (lane & 15);
  const float sc = scale[n], sh = shift[n];
  for (int mt = 0; mt < 13; ++mt)
    for (int r = 0; r < 8; ++r) {
      int p = mt * 16 + r + hiLane;
      if (p < LPOS)
        Y[(size_t)obl * CC * SP + (size_t)n * SP + t * LPOS + p] =
            (_Float16)bn_gelu(acc[mt][r], sc, sh);
    }
}

// =====================================================================
// Kernel 6: output projection GEMM (25088 x 64 @ 64 x 768) + bias, f32 out
// Block = 16 tokens x 768 outs; 4 waves split the 48 N-tiles.
// =====================================================================
__global__ void k_out_proj(const _Float16* __restrict__ y2, const float* __restrict__ out_w,
                           const float* __restrict__ out_b, float* __restrict__ out) {
  __shared__ __align__(16) _Float16 lA[16 * 64];
  const int tid = threadIdx.x, wave = tid >> 5, lane = tid & 31;
  const int tok0 = blockIdx.x * 16;
  for (int idx = tid; idx < 16 * 64; idx += 128) {
    int m = idx >> 6, c = idx & 63;
    int tok = tok0 + m;
    int p = tok >> 7, bt = tok & 127;
    int bbi = bt >> 3, tq = bt & 7;
    lA[m * 64 + c] = y2[((size_t)bbi * CC + c) * SP + tq * LPOS + p];
  }
  __syncthreads();
  const int hiLane = (lane >> 4) << 3;
  half16 a0 = frag_ld(lA, 64);
  half16 a1 = frag_ld(lA + 32, 64);
  for (int j = 0; j < 12; ++j) {
    int ntile = wave * 12 + j;
    int n = ntile * 16 + (lane & 15);
    float8 acc = {};
    for (int kc = 0; kc < 2; ++kc) {
      const float* wr = out_w + (size_t)n * DHC + kc * 32 + hiLane;
      half16 bfr;
      for (int e = 0; e < 8; ++e) bfr[e] = (_Float16)wr[e];
      for (int e = 0; e < 8; ++e) bfr[e + 8] = (_Float16)wr[16 + e];
      acc = wmma32(kc ? a1 : a0, bfr, acc);
    }
    for (int r = 0; r < 8; ++r) {
      int m = r + hiLane;
      out[(size_t)(tok0 + m) * DD + n] = acc[r] + out_b[n];
    }
  }
}

// =====================================================================
extern "C" void kernel_launch(void* const* d_in, const int* in_sizes, int n_in,
                              void* d_out, int out_size, void* d_ws, size_t ws_size,
                              hipStream_t stream) {
  (void)in_sizes; (void)n_in; (void)out_size; (void)ws_size;
  const float* x      = (const float*)d_in[0];
  const float* ln_g   = (const float*)d_in[1];
  const float* ln_b   = (const float*)d_in[2];
  const float* in_w   = (const float*)d_in[3];
  const float* in_b   = (const float*)d_in[4];
  const float* ext_w  = (const float*)d_in[5];
  const float* ext_sc = (const float*)d_in[6];
  const float* ext_sh = (const float*)d_in[7];
  const float* i0_w   = (const float*)d_in[8];
  const float* i0_sc  = (const float*)d_in[9];
  const float* i0_sh  = (const float*)d_in[10];
  const float* i1_w   = (const float*)d_in[11];
  const float* i1_sc  = (const float*)d_in[12];
  const float* i1_sh  = (const float*)d_in[13];
  const float* i2_w   = (const float*)d_in[14];
  const float* i2_sc  = (const float*)d_in[15];
  const float* i2_sh  = (const float*)d_in[16];
  const float* out_w  = (const float*)d_in[17];
  const float* out_b  = (const float*)d_in[18];
  float* out = (float*)d_out;

  char* ws = (char*)d_ws;
  size_t off = 0;
  auto alloc = [&](size_t bytes) {
    size_t o = off;
    off = (off + bytes + 255) & ~(size_t)255;
    return o;
  };
  float*    h  = (float*)(ws + alloc((size_t)NTOK * DHC * 4));
  _Float16* fn = (_Float16*)(ws + alloc((size_t)BT * POSP * DHC * 2));
  _Float16* s  = (_Float16*)(ws + alloc((size_t)80 * NCHP * SP * 2));
  _Float16* y0 = (_Float16*)(ws + alloc((size_t)80 * CC * SP * 2));
  _Float16* y1 = (_Float16*)(ws + alloc((size_t)80 * CC * SP * 2));
  _Float16* y2 = (_Float16*)(ws + alloc((size_t)16 * CC * SP * 2));

  k_ln_proj<<<NTOK / 64, 128, 0, stream>>>(x, ln_g, ln_b, in_w, in_b, h);
  k_normalize<<<(BT * POSP + 255) / 256, 256, 0, stream>>>(h, fn);
  k_corr<<<dim3(640, 4), 128, 0, stream>>>(fn, s);
  k_ext<<<dim3(80, 25), 128, 0, stream>>>(s, ext_w, ext_sc, ext_sh, y0);
  k_conv<<<dim3(80, 8), 128, 0, stream>>>(y0, i0_w, i0_sc, i0_sh, y1, 1);
  k_conv<<<dim3(80, 8), 128, 0, stream>>>(y1, i1_w, i1_sc, i1_sh, y0, 1);
  k_conv<<<dim3(16, 8), 128, 0, stream>>>(y0, i2_w, i2_sc, i2_sh, y2, 5);
  k_out_proj<<<NTOK / 16, 128, 0, stream>>>(y2, out_w, out_b, out);
}
